// AttentionBlock_41961830482587
// MI455X (gfx1250) — matrix operand order repaired and checked by reading, not compiled, not run on hardware
//
#include <hip/hip_runtime.h>

// ---------------------------------------------------------------------------
// Fused attention block for MI455X (gfx1250, wave32, WMMA 16x16x32 f16).
// mq/mk/mv = proj(q/k/v); scores = mq mk^T / 8; masked softmax; out = P mv.
// ---------------------------------------------------------------------------

typedef _Float16 v16h __attribute__((ext_vector_type(16)));
typedef _Float16 v8h  __attribute__((ext_vector_type(8)));
typedef _Float16 v4h  __attribute__((ext_vector_type(4)));
typedef float    v8f  __attribute__((ext_vector_type(8)));

union F16x16 { v16h v; v8h h[2]; };

constexpr int Bb = 2;
constexpr int S  = 2048;
constexpr int D  = 1024;
constexpr int H  = 16;
constexpr int HD = 64;

// Force fragments simultaneously live in distinct registers so the scheduler
// batches their loads (one wait) ahead of a WMMA chain.
__device__ __forceinline__ void pin4(v16h& a, v16h& b, v16h& c, v16h& d) {
    asm volatile("" : "+v"(a), "+v"(b), "+v"(c), "+v"(d));
}

// ---------------------------------------------------------------------------
// Projection GEMM:  Y[m][n] = sum_k X[m][k] * W[n][k] + bias[n]
// Block tile: 128(M) x 64(N), K-step 32. 8 waves; wave w owns rows [16w,16w+16).
// Output f16: TRANSPOSE_OUT==0 -> [B,H,S,HD]; ==1 -> [B,H,HD,S] (for V).
// ---------------------------------------------------------------------------
template <int TRANSPOSE_OUT>
__global__ __launch_bounds__(256)
void proj_gemm(const float* __restrict__ X, const float* __restrict__ W,
               const float* __restrict__ bias, _Float16* __restrict__ Y)
{
    __shared__ _Float16 As[128 * 32];   // [row][k] f16
    __shared__ _Float16 Ws[64 * 32];    // [n][k]  f16  (row-major W slice)

    const int tid  = threadIdx.x;
    const int wave = tid >> 5;
    const int lane = tid & 31;
    const int lh   = lane & 15;
    const int hi   = lane >> 4;
    const int m0   = blockIdx.y * 128;
    const int n0   = blockIdx.x * 64;

    v8f acc[4] = {};

    for (int k0 = 0; k0 < D; k0 += 32) {
        // ---- stage A tile (128x32 f32 -> f16), 4 float4 per thread ----
        #pragma unroll
        for (int i = 0; i < 4; ++i) {
            const int f   = tid + 256 * i;
            const int row = f >> 3;
            const int c4  = f & 7;
            const float4 x =
                reinterpret_cast<const float4*>(X + (size_t)(m0 + row) * D + k0)[c4];
            v4h hx;
            hx.x = (_Float16)x.x; hx.y = (_Float16)x.y;
            hx.z = (_Float16)x.z; hx.w = (_Float16)x.w;
            *reinterpret_cast<v4h*>(As + row * 32 + c4 * 4) = hx;
        }
        // ---- stage W tile (64x32 f32 -> f16), 2 float4 per thread ----
        #pragma unroll
        for (int i = 0; i < 2; ++i) {
            const int f   = tid + 256 * i;
            const int row = f >> 3;
            const int c4  = f & 7;
            const float4 x =
                reinterpret_cast<const float4*>(W + (size_t)(n0 + row) * D + k0)[c4];
            v4h hx;
            hx.x = (_Float16)x.x; hx.y = (_Float16)x.y;
            hx.z = (_Float16)x.z; hx.w = (_Float16)x.w;
            *reinterpret_cast<v4h*>(Ws + row * 32 + c4 * 4) = hx;
        }
        __syncthreads();

        // ---- A fragment (16x32)
        const _Float16* arow = As + (wave * 16 + lh) * 32;
        F16x16 a;
        a.h[0] = *reinterpret_cast<const v8h*>(arow + hi * 8);
        a.h[1] = *reinterpret_cast<const v8h*>(arow + 16 + hi * 8);

        // ---- batch all 4 B fragments (distinct regs), then 4 WMMAs ----
        F16x16 bm[4];
        #pragma unroll
        for (int nf = 0; nf < 4; ++nf) {
            const _Float16* brow = Ws + (nf * 16 + lh) * 32 + hi * 8;
            bm[nf].h[0] = *reinterpret_cast<const v8h*>(brow);
            bm[nf].h[1] = *reinterpret_cast<const v8h*>(brow + 16);
        }
        pin4(bm[0].v, bm[1].v, bm[2].v, bm[3].v);
        #pragma unroll
        for (int nf = 0; nf < 4; ++nf) {
            acc[nf] = __builtin_amdgcn_wmma_f32_16x16x32_f16(
                false, a.v, false, bm[nf].v, (short)0, acc[nf], false, false);
        }
        __syncthreads();
    }

    // ---- epilogue: bias, f16 convert, head-split scatter ----
    #pragma unroll
    for (int nf = 0; nf < 4; ++nf) {
        const int   ncol = n0 + nf * 16 + lh;
        const float bv   = bias[ncol];
        const int   h    = ncol >> 6;
        const int   d    = ncol & (HD - 1);
        #pragma unroll
        for (int r = 0; r < 8; ++r) {
            const int m  = m0 + wave * 16 + r + hi * 8;
            const int bb = m >> 11;
            const int s  = m & (S - 1);
            const float val = acc[nf][r] + bv;
            const size_t idx = TRANSPOSE_OUT
                ? ((size_t)((bb * H + h) * HD + d)) * S + s     // [B,H,HD,S]
                : ((size_t)((bb * H + h) * S + s)) * HD + d;    // [B,H,S,HD]
            Y[idx] = (_Float16)val;
        }
    }
}

// ---------------------------------------------------------------------------
// Fused flash attention. Grid: (S/128, B*H). Block 256 = 8 waves; each wave
// owns 16 q rows, streams 32-key tiles. WMMA fragments loaded straight from
// global (b128, L2-resident K/V); P transposed C->A layout via per-wave LDS.
// Row-sums of P computed with a WMMA against a ones fragment (replaces the
// 32-op shuffle-add butterfly per tile with one matrix op).
// ---------------------------------------------------------------------------
__global__ __launch_bounds__(256)
void attn_fused(const _Float16* __restrict__ mq, const _Float16* __restrict__ mk,
                const _Float16* __restrict__ mvT, const unsigned char* __restrict__ mask,
                float* __restrict__ out)
{
    __shared__ _Float16 Ps[8][16 * 32];
    __shared__ __align__(8) unsigned char maskLds[S];

    const int tid  = threadIdx.x;
    const int wave = tid >> 5;
    const int lane = tid & 31;
    const int lh   = lane & 15;
    const int hi   = lane >> 4;

    const int bh = blockIdx.y;
    const int b  = bh >> 4;
    const int h  = bh & 15;
    const _Float16* qp = mq  + (size_t)bh * S * HD;
    const _Float16* kp = mk  + (size_t)bh * S * HD;
    const _Float16* vp = mvT + (size_t)bh * HD * S;
    const unsigned char* mrow = mask + b * S;   // jax bool -> 1 byte
    const int q0 = blockIdx.x * 128 + wave * 16;

    // stage this batch's key mask row into LDS (2 KB, 8 B per thread)
    reinterpret_cast<unsigned long long*>(maskLds)[tid] =
        reinterpret_cast<const unsigned long long*>(mrow)[tid];
    __syncthreads();

    // Q A-fragments, pre-scaled by 1/sqrt(HD)=0.125 (folds the score scaling)
    F16x16 aq[2];
    {
        const _Float16* qrow = qp + (size_t)(q0 + lh) * HD;
        aq[0].h[0] = *reinterpret_cast<const v8h*>(qrow + hi * 8);
        aq[0].h[1] = *reinterpret_cast<const v8h*>(qrow + 16 + hi * 8);
        aq[1].h[0] = *reinterpret_cast<const v8h*>(qrow + 32 + hi * 8);
        aq[1].h[1] = *reinterpret_cast<const v8h*>(qrow + 48 + hi * 8);
        const _Float16 sc16 = (_Float16)0.125f;
        aq[0].v *= sc16;
        aq[1].v *= sc16;
    }

    // all-ones B fragment: P x ones -> row sums of P in C layout
    F16x16 ones;
    #pragma unroll
    for (int i = 0; i < 16; ++i) ones.v[i] = (_Float16)1.0f;

    float m_run[8], l_run[8];
    #pragma unroll
    for (int r = 0; r < 8; ++r) { m_run[r] = -3.0e38f; l_run[r] = 0.0f; }
    v8f acc[4] = {};

    for (int kt = 0; kt < S / 32; ++kt) {
        const int j0 = kt * 32;
        if (kt + 1 < S / 32) {   // hint next key/value tile into cache
            __builtin_prefetch(kp + (size_t)(j0 + 32 + lh) * HD, 0, 1);
            __builtin_prefetch(vp + (size_t)lh * S + j0 + 32, 0, 1);
        }

        // ---- batch all 4 K B-fragments (distinct regs), 4 score WMMAs ----
        F16x16 bk[2][2];
        #pragma unroll
        for (int ks = 0; ks < 2; ++ks) {
            #pragma unroll
            for (int nf = 0; nf < 2; ++nf) {
                const _Float16* krow =
                    kp + (size_t)(j0 + nf * 16 + lh) * HD + ks * 32 + hi * 8;
                bk[ks][nf].h[0] = *reinterpret_cast<const v8h*>(krow);
                bk[ks][nf].h[1] = *reinterpret_cast<const v8h*>(krow + 16);
            }
        }
        pin4(bk[0][0].v, bk[0][1].v, bk[1][0].v, bk[1][1].v);
        v8f sc[2];
        sc[0] = __builtin_amdgcn_wmma_f32_16x16x32_f16(
            false, aq[0].v, false, bk[0][0].v, (short)0, v8f{}, false, false);
        sc[1] = __builtin_amdgcn_wmma_f32_16x16x32_f16(
            false, aq[0].v, false, bk[0][1].v, (short)0, v8f{}, false, false);
        sc[0] = __builtin_amdgcn_wmma_f32_16x16x32_f16(
            false, aq[1].v, false, bk[1][0].v, (short)0, sc[0], false, false);
        sc[1] = __builtin_amdgcn_wmma_f32_16x16x32_f16(
            false, aq[1].v, false, bk[1][1].v, (short)0, sc[1], false, false);

        // ---- issue V B-fragment loads NOW (no pin: a pin would force an
        //      immediate loadcnt-0 wait; they stay live across the softmax
        //      and complete under the VALU work below) ----
        F16x16 bv[4];
        #pragma unroll
        for (int df = 0; df < 4; ++df) {
            const _Float16* vrow = vp + (size_t)(df * 16 + lh) * S + j0 + hi * 8;
            bv[df].h[0] = *reinterpret_cast<const v8h*>(vrow);
            bv[df].h[1] = *reinterpret_cast<const v8h*>(vrow + 16);
        }

        const float kv0 = maskLds[j0 + lh]      ? 1.0f : 0.0f;   // column mask
        const float kv1 = maskLds[j0 + 16 + lh] ? 1.0f : 0.0f;

        _Float16* pw = &Ps[wave][0];
        float sold[8];
        #pragma unroll
        for (int r = 0; r < 8; ++r) {
            // row-wise max over 32 keys (lane halves hold distinct q rows)
            float vmax = fmaxf(sc[0][r], sc[1][r]);
            #pragma unroll
            for (int off = 8; off >= 1; off >>= 1)
                vmax = fmaxf(vmax, __shfl_xor(vmax, off, 16));
            const float mnew = fmaxf(m_run[r], vmax);
            sold[r]  = __expf(m_run[r] - mnew);
            m_run[r] = mnew;
            const float p0 = __expf(sc[0][r] - mnew) * kv0;  // masked cols -> exact 0
            const float p1 = __expf(sc[1][r] - mnew) * kv1;
            // C-layout -> LDS row-major P tile (f16)
            pw[(r + hi * 8) * 32 + lh]      = (_Float16)p0;
            pw[(r + hi * 8) * 32 + 16 + lh] = (_Float16)p1;
            #pragma unroll
            for (int df = 0; df < 4; ++df) acc[df][r] *= sold[r];
        }

        asm volatile("s_wait_dscnt 0x0" ::: "memory");  // P stores visible before reload

        // ---- P as A-fragment (16x32) ----
        F16x16 ap;
        {
            const _Float16* pr = &Ps[wave][lh * 32];
            ap.h[0] = *reinterpret_cast<const v8h*>(pr + hi * 8);
            ap.h[1] = *reinterpret_cast<const v8h*>(pr + 16 + hi * 8);
        }
        // row sums of P via matrix op (every output column = rowsum)
        v8f lsum = __builtin_amdgcn_wmma_f32_16x16x32_f16(
            false, ap.v, false, ones.v, (short)0, v8f{}, false, false);
        // 4 independent PV WMMAs
        #pragma unroll
        for (int df = 0; df < 4; ++df) {
            acc[df] = __builtin_amdgcn_wmma_f32_16x16x32_f16(
                false, ap.v, false, bv[df].v, (short)0, acc[df], false, false);
        }
        #pragma unroll
        for (int r = 0; r < 8; ++r)
            l_run[r] = l_run[r] * sold[r] + lsum[r];
    }

    // ---- normalize, apply row mask, write [B,S,H*HD] f32
    #pragma unroll
    for (int df = 0; df < 4; ++df) {
        const int d = df * 16 + lh;
        #pragma unroll
        for (int r = 0; r < 8; ++r) {
            const int   qrow = q0 + r + hi * 8;
            const float qm   = maskLds[qrow] ? 1.0f : 0.0f;
            const float inv  = qm / fmaxf(l_run[r], 1e-30f);
            out[((size_t)(b * S + qrow)) * D + h * HD + d] = acc[df][r] * inv;
        }
    }
}

// ---------------------------------------------------------------------------
extern "C" void kernel_launch(void* const* d_in, const int* in_sizes, int n_in,
                              void* d_out, int out_size, void* d_ws, size_t ws_size,
                              hipStream_t stream)
{
    const float* q  = (const float*)d_in[0];
    const float* k  = (const float*)d_in[1];
    const float* v  = (const float*)d_in[2];
    const unsigned char* mask = (const unsigned char*)d_in[3];
    const float* Wq = (const float*)d_in[4];
    const float* bq = (const float*)d_in[5];
    const float* Wk = (const float*)d_in[6];
    const float* bk = (const float*)d_in[7];
    const float* Wv = (const float*)d_in[8];
    const float* bv = (const float*)d_in[9];
    float* out = (float*)d_out;

    _Float16* ws = (_Float16*)d_ws;
    const size_t per = (size_t)Bb * H * S * HD;      // 4,194,304 halfs = 8 MB
    _Float16* mq  = ws;                               // [B,H,S,HD]
    _Float16* mkh = ws + per;                         // [B,H,S,HD]
    _Float16* mvT = ws + 2 * per;                     // [B,H,HD,S]

    dim3 pgrid(D / 64, (Bb * S) / 128);               // (16, 32)
    proj_gemm<0><<<pgrid, 256, 0, stream>>>(q, Wq, bq, mq);
    proj_gemm<0><<<pgrid, 256, 0, stream>>>(k, Wk, bk, mkh);
    proj_gemm<1><<<pgrid, 256, 0, stream>>>(v, Wv, bv, mvT);

    dim3 agrid(S / 128, Bb * H);                      // (16, 32)
    attn_fused<<<agrid, 256, 0, stream>>>(mq, mkh, mvT, mask, out);
}
